// CRFasRNN_41979010351122
// MI455X (gfx1250) — matrix operand order repaired and checked by reading, not compiled, not run on hardware
//
#include <hip/hip_runtime.h>
#include <hip/hip_bf16.h>
#include <hip/hip_fp16.h>

// ---------------------------------------------------------------------------
// CRF-as-RNN on MI455X (gfx1250, wave32).
//   N = 6400 pixels, C = 2 classes, 5 mean-field iterations.
//   K_sp / K_bi materialized once as FP16 (164 MB total -> resident in 192MB
//   L2), each iteration does K @ q via V_WMMA_F32_16X16X32_F16.
//   q is pre-packed into the exact B-fragment lane layout so the WMMA inner
//   loop is branch-free: 6 coalesced b128 loads + 2 WMMAs per 32-wide chunk.
// ---------------------------------------------------------------------------

#define HH 80
#define WW 80
#define NPIX (HH * WW)          // 6400
#define NCHUNK (NPIX / 32)      // 200 K-dim chunks of 32
#define NITER 5

typedef __attribute__((ext_vector_type(8)))  _Float16 v8h;
typedef __attribute__((ext_vector_type(16))) _Float16 v16h;
typedef __attribute__((ext_vector_type(8)))  float    v8f;

// Scatter q (2 classes) for pixel i into the packed B-fragment image:
// qPad[chunk][lane][elem], lane = class + (k>=16)*16, elem = k&15,
// where k = i % 32 is the K-row within the chunk. Lanes 2..15 / 18..31
// stay zero (written once by crf_zero_pad).
__device__ __forceinline__ void pack_q(_Float16* __restrict__ qPad, int i,
                                       float q0, float q1) {
    int chunk = i >> 5;
    int k     = i & 31;
    int laneB = k & 16;            // 0 or 16
    int elem  = k & 15;
    _Float16* base = qPad + ((size_t)chunk * 32 + laneB) * 16 + elem;
    base[0]  = (_Float16)q0;       // lane laneB+0  : class 0 column
    base[16] = (_Float16)q1;       // lane laneB+1  : class 1 column
}

// ---------------------------------------------------------------------------
// Kernel 0: zero the packed-q buffer (200 chunks * 32 lanes * 16 f16).
// ---------------------------------------------------------------------------
__global__ void crf_zero_pad(uint4* __restrict__ qPad4) {
    int t = blockIdx.x * blockDim.x + threadIdx.x;
    const int n4 = NCHUNK * 32 * 16 * 2 / 16;  // 204800 B / 16
    if (t < n4) qPad4[t] = make_uint4(0u, 0u, 0u, 0u);
}

// ---------------------------------------------------------------------------
// Kernel 1: unaries, initial q = softmax(u), packed features.
// feat[i][0..7] = { y/3, x/3, y/80, x/80, r/13, g/13, b/13, 0 }
// ---------------------------------------------------------------------------
__global__ void crf_init_kernel(const float* __restrict__ image,
                                const float* __restrict__ rgb,
                                const int*   __restrict__ anno,
                                float* __restrict__ u,        // [N][2]
                                float* __restrict__ feat,     // [N][8]
                                _Float16* __restrict__ qPad,  // packed B image
                                float* __restrict__ qf) {     // [N][2] fp32
    int i = blockIdx.x * blockDim.x + threadIdx.x;
    if (i >= NPIX) return;

    const float PE = 0.69314718056f;   // -log(0.5)
    const float NE = 1.38629436112f;   // -log(0.25)
    float img = image[i];
    int   a   = anno[i];
    float d0 = (img - 10.0f)  * (img - 10.0f)  * 1e-4f;
    float d1 = (img - 120.0f) * (img - 120.0f) * 1e-4f;
    float u0 = (a == 0) ? (PE + d0) : ((a == 1) ? (NE + d0) : 0.0f);
    float u1 = (a == 0) ? (NE + d1) : ((a == 1) ? (PE + d1) : 0.0f);
    u[2 * i + 0] = u0;
    u[2 * i + 1] = u1;

    // q = softmax(u)  (reference applies softmax to u directly)
    float mx = fmaxf(u0, u1);
    float e0 = __expf(u0 - mx), e1 = __expf(u1 - mx);
    float inv = 1.0f / (e0 + e1);
    float q0 = e0 * inv, q1 = e1 * inv;
    pack_q(qPad, i, q0, q1);
    qf[2 * i + 0] = q0;
    qf[2 * i + 1] = q1;

    float y = (float)(i / WW), x = (float)(i % WW);
    const float ig = 1.0f / 3.0f, ia = 1.0f / 80.0f, ib = 1.0f / 13.0f;
    float* f = feat + (size_t)i * 8;
    f[0] = y * ig;  f[1] = x * ig;
    f[2] = y * ia;  f[3] = x * ia;
    f[4] = rgb[0 * NPIX + i] * ib;
    f[5] = rgb[1 * NPIX + i] * ib;
    f[6] = rgb[2 * NPIX + i] * ib;
    f[7] = 0.0f;
}

// ---------------------------------------------------------------------------
// Kernel 2: build K_sp, K_bi (fp16) and row normalizers. One block per row i.
// ---------------------------------------------------------------------------
__global__ void crf_buildK_kernel(const float* __restrict__ feat,
                                  _Float16* __restrict__ Ksp,
                                  _Float16* __restrict__ Kbi,
                                  float* __restrict__ n_sp,
                                  float* __restrict__ n_bi) {
    __shared__ float ssp[256];
    __shared__ float sbi[256];
    int i   = blockIdx.x;
    int tid = threadIdx.x;

    float4 fiA = *(const float4*)(feat + (size_t)i * 8 + 0);
    float4 fiB = *(const float4*)(feat + (size_t)i * 8 + 4);
    float accs = 0.0f, accb = 0.0f;

    for (int j = tid; j < NPIX; j += 256) {
        float4 fjA = *(const float4*)(feat + (size_t)j * 8 + 0);
        float4 fjB = *(const float4*)(feat + (size_t)j * 8 + 4);
        float d;
        float dsp = 0.0f;
        d = fiA.x - fjA.x; dsp = fmaf(d, d, dsp);
        d = fiA.y - fjA.y; dsp = fmaf(d, d, dsp);
        float dbi = 0.0f;
        d = fiA.z - fjA.z; dbi = fmaf(d, d, dbi);
        d = fiA.w - fjA.w; dbi = fmaf(d, d, dbi);
        d = fiB.x - fjB.x; dbi = fmaf(d, d, dbi);
        d = fiB.y - fjB.y; dbi = fmaf(d, d, dbi);
        d = fiB.z - fjB.z; dbi = fmaf(d, d, dbi);
        float ksp = __expf(-0.5f * dsp);
        float kbi = __expf(-0.5f * dbi);
        Ksp[(size_t)i * NPIX + j] = (_Float16)ksp;
        Kbi[(size_t)i * NPIX + j] = (_Float16)kbi;
        accs += ksp;
        accb += kbi;
    }
    ssp[tid] = accs;
    sbi[tid] = accb;
    __syncthreads();
    for (int s = 128; s > 0; s >>= 1) {
        if (tid < s) { ssp[tid] += ssp[tid + s]; sbi[tid] += sbi[tid + s]; }
        __syncthreads();
    }
    if (tid == 0) { n_sp[i] = ssp[0]; n_bi[i] = sbi[0]; }
}

// ---------------------------------------------------------------------------
// Kernel 3: sp = (Ksp@q)/n_sp, bi = (Kbi@q)/n_bi via WMMA f32_16x16x32_f16.
// One wave32 per 16-row output tile. B fragments come pre-packed from qPad
// (one unconditional 32B load per lane per chunk). Inner loop is manually
// unrolled x2 so the second set's loads overlap the first set's WMMAs.
// ---------------------------------------------------------------------------
__global__ void crf_gemm_kernel(const _Float16* __restrict__ Ksp,
                                const _Float16* __restrict__ Kbi,
                                const _Float16* __restrict__ qPad,
                                const float* __restrict__ n_sp,
                                const float* __restrict__ n_bi,
                                float* __restrict__ sp,            // [N][2]
                                float* __restrict__ bi) {          // [N][2]
    int gtid = blockIdx.x * blockDim.x + threadIdx.x;
    int tile = gtid >> 5;                 // wave index == 16-row tile
    if (tile >= NPIX / 16) return;        // uniform per-wave
    int lane = threadIdx.x & 31;
    int l15  = lane & 15;                 // A row within tile / B,D column
    int half = lane >> 4;                 // 0 or 1
    int i0   = tile * 16;

    v8f accS = {};
    v8f accB = {};

    // A fragment row base for this lane (K-offsets: half==0 -> {0..15},
    // half==1 -> {16..31}); two 16B loads per chunk per matrix.
    const _Float16* rowS = Ksp  + (size_t)(i0 + l15) * NPIX + half * 16;
    const _Float16* rowB = Kbi  + (size_t)(i0 + l15) * NPIX + half * 16;
    const _Float16* bptr = qPad + (size_t)lane * 16;   // +c*512 per chunk

    union AF { v16h v; v8h h[2]; };

#pragma unroll 1
    for (int c = 0; c < NCHUNK; c += 2) {
        int j0 = c * 32, j1 = j0 + 32;
        AF aS0, aB0, aS1, aB1;
        v16h b0, b1;
        // Issue all 12 loads for two chunks up front.
        aS0.h[0] = *(const v8h*)(rowS + j0);
        aS0.h[1] = *(const v8h*)(rowS + j0 + 8);
        aB0.h[0] = *(const v8h*)(rowB + j0);
        aB0.h[1] = *(const v8h*)(rowB + j0 + 8);
        b0       = *(const v16h*)(bptr + (size_t)c * 512);
        aS1.h[0] = *(const v8h*)(rowS + j1);
        aS1.h[1] = *(const v8h*)(rowS + j1 + 8);
        aB1.h[0] = *(const v8h*)(rowB + j1);
        aB1.h[1] = *(const v8h*)(rowB + j1 + 8);
        b1       = *(const v16h*)(bptr + (size_t)(c + 1) * 512);

        accS = __builtin_amdgcn_wmma_f32_16x16x32_f16(
            false, aS0.v, false, b0, (short)0, accS, false, false);
        accB = __builtin_amdgcn_wmma_f32_16x16x32_f16(
            false, aB0.v, false, b0, (short)0, accB, false, false);
        accS = __builtin_amdgcn_wmma_f32_16x16x32_f16(
            false, aS1.v, false, b1, (short)0, accS, false, false);
        accB = __builtin_amdgcn_wmma_f32_16x16x32_f16(
            false, aB1.v, false, b1, (short)0, accB, false, false);
    }

    // D layout: VGPR r -> row r + 8*half, lane -> column (= class for col<2).
    if (l15 < 2) {
        int rbase = half * 8;
#pragma unroll
        for (int r = 0; r < 8; ++r) {
            int i = i0 + rbase + r;
            sp[2 * i + l15] = accS[r] / n_sp[i];
            bi[2 * i + l15] = accB[r] / n_bi[i];
        }
    }
}

// ---------------------------------------------------------------------------
// Kernel 4: per-pixel compatibility transform + softmax update of q.
// ---------------------------------------------------------------------------
__global__ void crf_update_kernel(const float* __restrict__ u,
                                  const float* __restrict__ sp,
                                  const float* __restrict__ bi,
                                  const float* __restrict__ wsp,
                                  const float* __restrict__ bsp,
                                  const float* __restrict__ wbi,
                                  const float* __restrict__ bbi,
                                  const float* __restrict__ wc,
                                  const float* __restrict__ bc,
                                  _Float16* __restrict__ qPad,
                                  float* __restrict__ qf) {
    int i = blockIdx.x * blockDim.x + threadIdx.x;
    if (i >= NPIX) return;
    float s0 = sp[2 * i], s1 = sp[2 * i + 1];
    float t0 = bi[2 * i], t1 = bi[2 * i + 1];
    // msg = sp @ w_sp^T + b_sp + bi @ w_bi^T + b_bi   (w row-major [C][C])
    float m0 = s0 * wsp[0] + s1 * wsp[1] + bsp[0] + t0 * wbi[0] + t1 * wbi[1] + bbi[0];
    float m1 = s0 * wsp[2] + s1 * wsp[3] + bsp[1] + t0 * wbi[2] + t1 * wbi[3] + bbi[1];
    float p0 = m0 * wc[0] + m1 * wc[1] + bc[0];
    float p1 = m0 * wc[2] + m1 * wc[3] + bc[1];
    float z0 = u[2 * i] - p0, z1 = u[2 * i + 1] - p1;
    float mx = fmaxf(z0, z1);
    float e0 = __expf(z0 - mx), e1 = __expf(z1 - mx);
    float inv = 1.0f / (e0 + e1);
    float q0 = e0 * inv, q1 = e1 * inv;
    pack_q(qPad, i, q0, q1);
    qf[2 * i]     = q0;
    qf[2 * i + 1] = q1;
}

// ---------------------------------------------------------------------------
// Kernel 5: nan_to_num + Threshold(0.5, 0), emit [1, C, H, W].
// ---------------------------------------------------------------------------
__global__ void crf_final_kernel(const float* __restrict__ qf,
                                 float* __restrict__ out) {
    int i = blockIdx.x * blockDim.x + threadIdx.x;
    if (i >= NPIX) return;
#pragma unroll
    for (int c = 0; c < 2; ++c) {
        float v = qf[2 * i + c];
        if (!(v == v)) v = 0.0f;          // nan -> 0
        out[c * NPIX + i] = (v > 0.5f) ? v : 0.0f;
    }
}

// ---------------------------------------------------------------------------
extern "C" void kernel_launch(void* const* d_in, const int* in_sizes, int n_in,
                              void* d_out, int out_size, void* d_ws, size_t ws_size,
                              hipStream_t stream) {
    const float* image = (const float*)d_in[0];   // [1,80,80]
    const float* rgb   = (const float*)d_in[1];   // [1,3,80,80]
    const int*   anno  = (const int*)  d_in[2];   // [80,80]
    const float* wsp   = (const float*)d_in[3];
    const float* bsp   = (const float*)d_in[4];
    const float* wbi   = (const float*)d_in[5];
    const float* bbi   = (const float*)d_in[6];
    const float* wc    = (const float*)d_in[7];
    const float* bc    = (const float*)d_in[8];
    float* out = (float*)d_out;

    // Carve workspace (256B-aligned blocks).
    char* ws = (char*)d_ws;
    size_t off = 0;
    auto carve = [&](size_t bytes) -> char* {
        char* p = ws + off;
        off += (bytes + 255) & ~(size_t)255;
        return p;
    };
    _Float16* Ksp  = (_Float16*)carve((size_t)NPIX * NPIX * sizeof(_Float16)); // 81.92MB
    _Float16* Kbi  = (_Float16*)carve((size_t)NPIX * NPIX * sizeof(_Float16)); // 81.92MB
    float*    feat = (float*)   carve((size_t)NPIX * 8 * sizeof(float));
    float*    u    = (float*)   carve((size_t)NPIX * 2 * sizeof(float));
    _Float16* qPad = (_Float16*)carve((size_t)NCHUNK * 32 * 16 * sizeof(_Float16)); // 200KB
    float*    qf   = (float*)   carve((size_t)NPIX * 2 * sizeof(float));
    float*    nsp  = (float*)   carve((size_t)NPIX * sizeof(float));
    float*    nbi  = (float*)   carve((size_t)NPIX * sizeof(float));
    float*    sp   = (float*)   carve((size_t)NPIX * 2 * sizeof(float));
    float*    bi   = (float*)   carve((size_t)NPIX * 2 * sizeof(float));

    const int T = 256;
    const int PB = (NPIX + T - 1) / T;   // 25 blocks for per-pixel kernels

    const int nzero4 = NCHUNK * 32 * 16 * 2 / 16;
    crf_zero_pad<<<(nzero4 + T - 1) / T, T, 0, stream>>>((uint4*)qPad);
    crf_init_kernel<<<PB, T, 0, stream>>>(image, rgb, anno, u, feat, qPad, qf);
    crf_buildK_kernel<<<NPIX, T, 0, stream>>>(feat, Ksp, Kbi, nsp, nbi);

    const int NTILE = NPIX / 16;                 // 400 waves
    const int GB = (NTILE * 32 + T - 1) / T;     // 50 blocks of 8 waves
    for (int it = 0; it < NITER; ++it) {
        crf_gemm_kernel<<<GB, T, 0, stream>>>(Ksp, Kbi, qPad, nsp, nbi, sp, bi);
        crf_update_kernel<<<PB, T, 0, stream>>>(u, sp, bi, wsp, bsp, wbi, bbi,
                                                wc, bc, qPad, qf);
    }
    crf_final_kernel<<<PB, T, 0, stream>>>(qf, out);
}
